// SAM_2903397892217
// MI455X (gfx1250) — matrix operand order repaired and checked by reading, not compiled, not run on hardware
//
#include <hip/hip_runtime.h>
#include <hip/hip_bf16.h>

typedef __attribute__((ext_vector_type(16))) _Float16 v16h;
typedef __attribute__((ext_vector_type(8)))  _Float16 v8h;
typedef __attribute__((ext_vector_type(8)))  float    v8f;

static constexpr int Bn = 4, Cn = 64, Hn = 96, Wn = 288;
static constexpr int HW = Hn * Wn;                   // 27648
static constexpr long CHW = (long)Cn * HW;           // 1,769,472
static constexpr long BCHW = (long)Bn * CHW;         // 7,077,888
static constexpr int BH = Bn * Hn;                   // 384
static constexpr int Pn = Bn * Hn * Wn;              // 110,592 pixels

// ---------------- WMMA helpers (CDNA5 wave32 fragment layouts) ----------------

__device__ __forceinline__ v8f wmma_f16(v16h a, v16h b, v8f c) {
  return __builtin_amdgcn_wmma_f32_16x16x32_f16(false, a, false, b, (short)0, c, false, false);
}

__device__ __forceinline__ v16h cat8(v8h lo, v8h hi) {
  v16h r;
#pragma unroll
  for (int i = 0; i < 8; ++i) { r[i] = lo[i]; r[i + 8] = hi[i]; }
  return r;
}

// A 16x32 f16 fragment, per-lane: 16 contiguous K halves starting at k0+16g
__device__ __forceinline__ v16h a_frag_f16(const _Float16* row, int k0, int g) {
  return *(const v16h*)(row + k0 + 16 * g);
}

// B 32x16 f16 fragment, per-lane: 16 contiguous K halves starting at k0+16g
__device__ __forceinline__ v16h b_frag_f16(const _Float16* col, int k0, int g) {
  return *(const v16h*)(col + k0 + 16 * g);
}

__device__ __forceinline__ v16h a_frag_f32(const float* row, int k0, int g) {
  v16h r;
  const float* p0 = row + k0 + 16 * g;
  const float* p1 = row + k0 + 16 * g + 8;
#pragma unroll
  for (int j = 0; j < 8; ++j) { r[j] = (_Float16)p0[j]; r[j + 8] = (_Float16)p1[j]; }
  return r;
}

__device__ __forceinline__ v16h b_frag_f32(const float* col, int k0, int g) {
  v16h r;
  const float* p = col + k0 + 16 * g;
#pragma unroll
  for (int j = 0; j < 16; ++j) r[j] = (_Float16)p[j];
  return r;
}

// ---------------- 3x3 conv (implicit GEMM, WMMA f16) ----------------
// out[b,o,h,w] = sum_{c,dy,dx} in[b,c,h+dy-1,w+dx-1] * wgt[o,c,dy,dx]  (+relu | +resid)
// Tile: 32 output pixels (along W) x 64 output channels per 128-thread block.
__global__ __launch_bounds__(128) void k_conv3(const float* __restrict__ in,
                                               const float* __restrict__ wgt,
                                               const float* __restrict__ resid,
                                               float* __restrict__ out,
                                               int do_relu) {
  __shared__ alignas(32) _Float16 sa[3 * 34 * 64];   // [dy][wp][c]  13,056 B
  __shared__ alignas(32) _Float16 sw3[3 * 64 * 64];  // [dx][o][i]   24,576 B (one dy group)
  __shared__ float sc[64 * 33];                      // C staging     8,448 B
  const int wt = blockIdx.x, h = blockIdx.y, b = blockIdx.z;
  const int w0 = wt * 32;
  const int tid = threadIdx.x;

  // activations with halo -> LDS f16, [dy][wp][c]
  for (int t = tid; t < 3 * 64 * 34; t += 128) {
    int dy = t / (64 * 34), rem = t % (64 * 34);
    int c = rem / 34, wp = rem % 34;
    int hh = h + dy - 1, ww = w0 + wp - 1;
    float v = 0.f;
    if ((unsigned)hh < (unsigned)Hn && (unsigned)ww < (unsigned)Wn)
      v = in[((long)(b * 64 + c) * Hn + hh) * Wn + ww];
    sa[(dy * 34 + wp) * 64 + c] = (_Float16)v;
  }

  const int wid = tid >> 5, lane = tid & 31, l16 = lane & 15, g = lane >> 4;
  const int n0 = wid * 16;
  v8f c0 = {}, c1 = {};

  for (int tg = 0; tg < 3; ++tg) {        // tg == dy
    __syncthreads();
    // weights for taps dy=tg, dx=0..2 -> LDS [dx][o][i]  (OIHW global)
    for (int t = tid; t < 3 * 64 * 64; t += 128) {
      int o = t / 192, rem = t % 192, i = rem / 3, dx = rem % 3;
      sw3[(dx * 64 + o) * 64 + i] = (_Float16)wgt[(o * 64 + i) * 9 + tg * 3 + dx];
    }
    __syncthreads();
#pragma unroll
    for (int dx = 0; dx < 3; ++dx) {
      const _Float16* bcol  = &sw3[(dx * 64 + (n0 + l16)) * 64];
      const _Float16* arow0 = &sa[(tg * 34 + (l16 + dx)) * 64];
      const _Float16* arow1 = &sa[(tg * 34 + (16 + l16 + dx)) * 64];
#pragma unroll
      for (int kk = 0; kk < 64; kk += 32) {
        v16h bf = b_frag_f16(bcol, kk, g);
        c0 = wmma_f16(a_frag_f16(arow0, kk, g), bf, c0);
        c1 = wmma_f16(a_frag_f16(arow1, kk, g), bf, c1);
      }
    }
  }
  // stage C tile to LDS [o][w] (pad 33) then store coalesced
#pragma unroll
  for (int r = 0; r < 8; ++r) {
    sc[(n0 + l16) * 33 + (r + 8 * g)]      = c0[r];
    sc[(n0 + l16) * 33 + (16 + r + 8 * g)] = c1[r];
  }
  __syncthreads();
  for (int t = tid; t < 2048; t += 128) {
    int o = t / 32, wl = t % 32;
    float v = sc[o * 33 + wl];
    long gi = ((long)(b * 64 + o) * Hn + h) * Wn + w0 + wl;
    if (do_relu) v = fmaxf(v, 0.f);
    else         v += resid[gi];
    out[gi] = v;
  }
}

// ---------------- 1x1 projection: out[pix,o] = sum_c in[b,c,h,w]*W[o,c] + bias[o] (f16 out)
__global__ __launch_bounds__(256) void k_proj(const float* __restrict__ in,
                                              const float* __restrict__ w,
                                              const float* __restrict__ bias,
                                              _Float16* __restrict__ outp) {
  __shared__ float sx[64 * 33];    // [c][w] pad 33
  __shared__ float swt[64 * 65];   // [o][c] pad 65 (bank-conflict free)
  __shared__ float sb[64];
  const int wt = blockIdx.x, h = blockIdx.y, b = blockIdx.z;
  const int w0 = wt * 32, tid = threadIdx.x;
  for (int t = tid; t < 2048; t += 256) {
    int c = t / 32, wl = t % 32;
    sx[c * 33 + wl] = in[((long)(b * 64 + c) * Hn + h) * Wn + w0 + wl];
  }
  for (int t = tid; t < 4096; t += 256) { int o = t / 64, c = t % 64; swt[o * 65 + c] = w[t]; }
  if (tid < 64) sb[tid] = bias[tid];
  __syncthreads();
  const long pixbase = ((long)(b * Hn + h) * Wn + w0);
#pragma unroll
  for (int i = 0; i < 8; ++i) {
    int e = tid + 256 * i;
    int o = e & 63, wl = e >> 6;
    float acc = sb[o];
    for (int c = 0; c < 64; ++c) acc += swt[o * 65 + c] * sx[c * 33 + wl];
    outp[(pixbase + wl) * 64 + o] = (_Float16)acc;
  }
}

// ---------------- score = Q(288x64) x S^T(64x288) per (b,h), WMMA f16 ----------------
__global__ __launch_bounds__(288) void k_score(const _Float16* __restrict__ Qh,
                                               const _Float16* __restrict__ Sh,
                                               float* __restrict__ raw) {
  const int tn = blockIdx.x, tm = blockIdx.y, bh = blockIdx.z;
  const int tid = threadIdx.x, wid = tid >> 5, lane = tid & 31;
  const int l16 = lane & 15, g = lane >> 4;
  const int mi = wid / 3, ni = wid % 3;
  const int wb = tm * 96 + mi * 32, vb = tn * 96 + ni * 32;
  v8f acc[2][2] = {};
  const _Float16* qbase = Qh + (long)bh * 288 * 64;
  const _Float16* sbase = Sh + (long)bh * 288 * 64;
#pragma unroll
  for (int kk = 0; kk < 64; kk += 32) {
    v16h a0 = a_frag_f16(qbase + (long)(wb + l16) * 64, kk, g);
    v16h a1 = a_frag_f16(qbase + (long)(wb + 16 + l16) * 64, kk, g);
    v16h b0 = b_frag_f16(sbase + (long)(vb + l16) * 64, kk, g);
    v16h b1 = b_frag_f16(sbase + (long)(vb + 16 + l16) * 64, kk, g);
    acc[0][0] = wmma_f16(a0, b0, acc[0][0]);
    acc[0][1] = wmma_f16(a0, b1, acc[0][1]);
    acc[1][0] = wmma_f16(a1, b0, acc[1][0]);
    acc[1][1] = wmma_f16(a1, b1, acc[1][1]);
  }
  float* obase = raw + (long)bh * 288 * 288;
#pragma unroll
  for (int mt = 0; mt < 2; ++mt)
#pragma unroll
    for (int nt = 0; nt < 2; ++nt)
#pragma unroll
      for (int r = 0; r < 8; ++r)
        obase[(long)(wb + 16 * mt + r + 8 * g) * 288 + vb + 16 * nt + l16] = acc[mt][nt][r];
}

// ---------------- softmax stats ----------------
__global__ __launch_bounds__(256) void k_rowstats(const float* __restrict__ raw,
                                                  float* __restrict__ rmax,
                                                  float* __restrict__ rinv) {
  const int row = blockIdx.x * 8 + (threadIdx.x >> 5);
  const int lane = threadIdx.x & 31;
  const float* p = raw + (long)row * 288;
  float s[9];
#pragma unroll
  for (int k = 0; k < 9; ++k) s[k] = p[lane + 32 * k];
  float m = s[0];
#pragma unroll
  for (int k = 1; k < 9; ++k) m = fmaxf(m, s[k]);
#pragma unroll
  for (int off = 16; off > 0; off >>= 1) m = fmaxf(m, __shfl_xor(m, off, 32));
  float l = 0.f;
#pragma unroll
  for (int k = 0; k < 9; ++k) l += __expf(s[k] - m);
#pragma unroll
  for (int off = 16; off > 0; off >>= 1) l += __shfl_xor(l, off, 32);
  if (lane == 0) { rmax[row] = m; rinv[row] = 1.f / l; }
}

__global__ __launch_bounds__(288) void k_colstats(const float* __restrict__ raw,
                                                  float* __restrict__ cmax,
                                                  float* __restrict__ cinv) {
  const int bh = blockIdx.x, v = threadIdx.x;
  const float* p = raw + (long)bh * 288 * 288 + v;
  float m = -3.4e38f, l = 0.f;
  for (int w = 0; w < 288; ++w) {
    float s = p[(long)w * 288];
    float nm = fmaxf(m, s);
    l = l * __expf(m - nm) + __expf(s - nm);
    m = nm;
  }
  cmax[bh * 288 + v] = m;
  cinv[bh * 288 + v] = 1.f / l;
}

// normalize: M1 <- row-softmax in place; M2[v,w] <- col-softmax (LDS transpose tile)
__global__ __launch_bounds__(256) void k_norm(float* __restrict__ M1,
                                              float* __restrict__ M2,
                                              const float* __restrict__ rmax,
                                              const float* __restrict__ rinv,
                                              const float* __restrict__ cmax,
                                              const float* __restrict__ cinv) {
  __shared__ float t2[32 * 33];
  const int v0 = blockIdx.x * 32, w0 = blockIdx.y * 32, bh = blockIdx.z;
  const int tid = threadIdx.x, vl = tid & 31, wl = tid >> 5;
  float* base1 = M1 + (long)bh * 288 * 288;
  float* base2 = M2 + (long)bh * 288 * 288;
#pragma unroll
  for (int i = 0; i < 4; ++i) {
    int w = w0 + wl + 8 * i, v = v0 + vl;
    long idx = (long)w * 288 + v;
    float s = base1[idx];
    base1[idx] = __expf(s - rmax[bh * 288 + w]) * rinv[bh * 288 + w];
    t2[(wl + 8 * i) * 33 + vl] = __expf(s - cmax[bh * 288 + v]) * cinv[bh * 288 + v];
  }
  __syncthreads();
#pragma unroll
  for (int i = 0; i < 4; ++i) {
    int vr = wl + 8 * i, wc = vl;
    base2[(long)(v0 + vr) * 288 + (w0 + wc)] = t2[wc * 33 + vr];
  }
}

// deterministic column sums: out[bh*288+v] = sum_w M[bh][w][v]
__global__ __launch_bounds__(288) void k_colsum(const float* __restrict__ M, float* __restrict__ outp) {
  const int bh = blockIdx.x, v = threadIdx.x;
  const float* p = M + (long)bh * 288 * 288 + v;
  float s = 0.f;
  for (int w = 0; w < 288; ++w) s += p[(long)w * 288];
  outp[bh * 288 + v] = s;
}

// ---------------- morphological close of invalid mask (disk radius 3, pad 6) ----------------
__global__ __launch_bounds__(256) void k_dilate(const float* __restrict__ vsum, float* __restrict__ dil) {
  int idx = blockIdx.x * 256 + threadIdx.x;
  if (idx >= Bn * 108 * 300) return;
  int b = idx / (108 * 300), rem = idx % (108 * 300);
  int y = rem / 300, x = rem % 300;
  float d = 0.f;
  for (int dy = -3; dy <= 3; ++dy)
    for (int dx = -3; dx <= 3; ++dx) {
      if (dy * dy + dx * dx > 9) continue;
      int my = y + dy - 6, mx = x + dx - 6;
      if ((unsigned)my < (unsigned)Hn && (unsigned)mx < (unsigned)Wn)
        if (vsum[((long)b * Hn + my) * Wn + mx] <= 0.1f) d = 1.f;
    }
  dil[idx] = d;
}

__global__ __launch_bounds__(256) void k_erode(const float* __restrict__ dil, float* __restrict__ V) {
  int idx = blockIdx.x * 256 + threadIdx.x;
  if (idx >= Pn) return;
  int b = idx / HW, rem = idx % HW;
  int h = rem / Wn, w = rem % Wn;
  int y = h + 6, x = w + 6;
  float e = 1.f;
  for (int dy = -3; dy <= 3; ++dy)
    for (int dx = -3; dx <= 3; ++dx) {
      if (dy * dy + dx * dx > 9) continue;
      if (dil[((long)b * 108 + (y + dy)) * 300 + (x + dx)] < 0.5f) e = 0.f;
    }
  V[idx] = 1.f - e;
}

// ---------------- warp GEMM: out[bh][row][c] = sum_v M[bh][row][v] * X[b,c,h,v] ----------------
__global__ __launch_bounds__(192) void k_warp(const float* __restrict__ M,
                                              const float* __restrict__ X,
                                              float* __restrict__ outbuf) {
  const int bh = blockIdx.x, b = bh / Hn, h = bh % Hn;
  const int tid = threadIdx.x, wid = tid >> 5, lane = tid & 31;
  const int l16 = lane & 15, g = lane >> 4;
  const int m0 = wid * 48;
  v8f acc[3][4] = {};
  const float* mbase = M + (long)bh * 288 * 288;
  const float* xbase = X + ((long)b * 64 * Hn + h) * Wn;   // + c*HW + v
  for (int k0 = 0; k0 < 288; k0 += 32) {
    v16h af[3];
#pragma unroll
    for (int mf = 0; mf < 3; ++mf)
      af[mf] = a_frag_f32(mbase + (long)(m0 + 16 * mf + l16) * 288, k0, g);
    v16h bf[4];
#pragma unroll
    for (int nf = 0; nf < 4; ++nf)
      bf[nf] = b_frag_f32(xbase + (long)(16 * nf + l16) * HW, k0, g);
#pragma unroll
    for (int mf = 0; mf < 3; ++mf)
#pragma unroll
      for (int nf = 0; nf < 4; ++nf)
        acc[mf][nf] = wmma_f16(af[mf], bf[nf], acc[mf][nf]);
  }
  float* obase = outbuf + (long)bh * 288 * 64;
#pragma unroll
  for (int mf = 0; mf < 3; ++mf)
#pragma unroll
    for (int nf = 0; nf < 4; ++nf)
#pragma unroll
      for (int r = 0; r < 8; ++r)
        obase[(long)(m0 + 16 * mf + r + 8 * g) * 64 + 16 * nf + l16] = acc[mf][nf][r];
}

// ---------------- final fusion: out[o] = bnw[o,:64].buf + bnw[o,64:128].x + bnw[o,128]*V + bnb[o]
__global__ __launch_bounds__(256) void k_out(const float* __restrict__ wbuf,
                                             const float* __restrict__ X,
                                             const float* __restrict__ V,
                                             const float* __restrict__ bnw,
                                             const float* __restrict__ bnb,
                                             float* __restrict__ outp) {
  __shared__ float sbn[64 * 129];
  __shared__ float sbuf[32 * 65];
  __shared__ float sx[64 * 33];
  __shared__ float sv[32];
  __shared__ float sbb[64];
  const int wt = blockIdx.x, h = blockIdx.y, b = blockIdx.z;
  const int w0 = wt * 32, tid = threadIdx.x;
  const int bh = b * Hn + h;
  for (int t = tid; t < 64 * 129; t += 256) sbn[t] = bnw[t];
  for (int t = tid; t < 2048; t += 256) {
    int wl = t >> 6, c = t & 63;
    sbuf[wl * 65 + c] = wbuf[((long)bh * 288 + w0 + wl) * 64 + c];
  }
  for (int t = tid; t < 2048; t += 256) {
    int c = t >> 5, wl = t & 31;
    sx[c * 33 + wl] = X[((long)(b * 64 + c) * Hn + h) * Wn + w0 + wl];
  }
  if (tid < 32) sv[tid] = V[(long)bh * Wn + w0 + tid];
  if (tid < 64) sbb[tid] = bnb[tid];
  __syncthreads();
#pragma unroll
  for (int i = 0; i < 8; ++i) {
    int e = tid + 256 * i;
    int wl = e & 31, o = e >> 5;
    float acc = sbb[o] + sbn[o * 129 + 128] * sv[wl];
    for (int c = 0; c < 64; ++c)
      acc += sbn[o * 129 + c] * sbuf[wl * 65 + c] + sbn[o * 129 + 64 + c] * sx[c * 33 + wl];
    outp[((long)(b * 64 + o) * Hn + h) * Wn + w0 + wl] = acc;
  }
}

// ---------------- launch ----------------
extern "C" void kernel_launch(void* const* d_in, const int* in_sizes, int n_in,
                              void* d_out, int out_size, void* d_ws, size_t ws_size,
                              hipStream_t stream) {
  const float* xL  = (const float*)d_in[0];
  const float* xR  = (const float*)d_in[1];
  const float* w1  = (const float*)d_in[2];
  const float* w2  = (const float*)d_in[3];
  const float* b1w = (const float*)d_in[4];
  const float* b1b = (const float*)d_in[5];
  const float* b2w = (const float*)d_in[6];
  const float* b2b = (const float*)d_in[7];
  const float* bnw = (const float*)d_in[8];
  const float* bnb = (const float*)d_in[9];

  float* out = (float*)d_out;
  float* oL = out;                                  // out_L
  float* oR = out + BCHW;                           // out_R
  float* M1 = out + 2 * BCHW;                       // M_r2l (raw score first, normalized in place)
  float* M2 = M1 + (long)BH * 288 * 288;            // M_l2r
  float* V1 = M2 + (long)BH * 288 * 288;            // V_r2l
  float* V2 = V1 + Pn;                              // V_l2r

  char* ws = (char*)d_ws;
  float*    tmp  = (float*)ws;                            // conv intermediate
  float*    bufL = (float*)(ws + (size_t)BCHW * 4);       // later reused as warpL
  float*    bufR = (float*)(ws + (size_t)BCHW * 8);       // later reused as warpR
  _Float16* Qh   = (_Float16*)(ws + (size_t)BCHW * 12);   // (b,h,w,c) f16
  _Float16* Sh   = (_Float16*)(ws + (size_t)BCHW * 14);   // (b,h,v,c) f16
  float* rmax = (float*)(ws + (size_t)BCHW * 16);
  float* rinv = rmax + Pn;
  float* cmax = rinv + Pn;
  float* cinv = cmax + Pn;
  float* vR2L = cinv + Pn;
  float* vL2R = vR2L + Pn;
  float* dilL = vL2R + Pn;
  float* dilR = dilL + (size_t)Bn * 108 * 300;
  float* warpL = bufL;
  float* warpR = bufR;

  dim3 gpix(Wn / 32, Hn, Bn);

  // residual blocks (implicit-GEMM WMMA conv)
  k_conv3<<<gpix, 128, 0, stream>>>(xL, w1, nullptr, tmp, 1);
  k_conv3<<<gpix, 128, 0, stream>>>(tmp, w2, xL, bufL, 0);
  k_conv3<<<gpix, 128, 0, stream>>>(xR, w1, nullptr, tmp, 1);
  k_conv3<<<gpix, 128, 0, stream>>>(tmp, w2, xR, bufR, 0);
  // Q / S projections (f16 channel-last outputs for WMMA fragments)
  k_proj<<<gpix, 256, 0, stream>>>(bufL, b1w, b1b, Qh);
  k_proj<<<gpix, 256, 0, stream>>>(bufR, b2w, b2b, Sh);
  // raw attention scores straight into the M_r2l output slot (L2-resident afterwards)
  k_score<<<dim3(3, 3, BH), 288, 0, stream>>>(Qh, Sh, M1);
  // bidirectional softmax
  k_rowstats<<<Pn / 8, 256, 0, stream>>>(M1, rmax, rinv);
  k_colstats<<<BH, 288, 0, stream>>>(M1, cmax, cinv);
  k_norm<<<dim3(9, 9, BH), 256, 0, stream>>>(M1, M2, rmax, rinv, cmax, cinv);
  // valid-mask sums (deterministic reductions)
  k_colsum<<<BH, 288, 0, stream>>>(M1, vR2L);
  k_colsum<<<BH, 288, 0, stream>>>(M2, vL2R);
  // morphological close -> V masks
  int nd = Bn * 108 * 300;
  k_dilate<<<(nd + 255) / 256, 256, 0, stream>>>(vL2R, dilL);
  k_dilate<<<(nd + 255) / 256, 256, 0, stream>>>(vR2L, dilR);
  k_erode<<<(Pn + 255) / 256, 256, 0, stream>>>(dilL, V2);
  k_erode<<<(Pn + 255) / 256, 256, 0, stream>>>(dilR, V1);
  // warped features (WMMA, M read back from L2)
  k_warp<<<BH, 192, 0, stream>>>(M1, xR, warpL);
  k_warp<<<BH, 192, 0, stream>>>(M2, xL, warpR);
  // final 129->64 fusion
  k_out<<<gpix, 256, 0, stream>>>(warpL, xL, V2, bnw, bnb, oL);
  k_out<<<gpix, 256, 0, stream>>>(warpR, xR, V1, bnw, bnb, oR);

  (void)in_sizes; (void)n_in; (void)out_size; (void)ws_size;
}
